// DINOv3Attention_77094662963469
// MI455X (gfx1250) — matrix operand order repaired and checked by reading, not compiled, not run on hardware
//
#include <hip/hip_runtime.h>
#include <hip/hip_bf16.h>

#define D_MODEL 768
#define NHEADS  12
#define HDIM    64
#define BATCH   8
#define SEQ     1025
#define SPAD    1056            // padded seq length, multiple of 32
#define MROWS   (BATCH*SEQ)     // 8200 token rows
#define QUARTER 16
#define ROPE_THETA 100.0f

typedef __bf16 bf16_t;
typedef __attribute__((ext_vector_type(16))) __bf16 v16bf;
typedef __attribute__((ext_vector_type(8)))  __bf16 v8bf;
typedef __attribute__((ext_vector_type(8)))  float   v8f;

static __device__ __forceinline__ v8f vzero() {
  v8f z;
#pragma unroll
  for (int i = 0; i < 8; ++i) z[i] = 0.0f;
  return z;
}

// A-operand (16x32 bf16, MxK), ISA 7.12.2:
// lanes 0-15 : row M=lane,    elems K{0..7} then K{16..23}
// lanes 16-31: row M=lane-16, elems K{8..15} then K{24..31}
static __device__ __forceinline__ v16bf load_a(const bf16_t* rowK0, int lane) {
  const int off = (lane & 16) ? 8 : 0;
  v8bf lo = *(const v8bf*)(rowK0 + off);
  v8bf hi = *(const v8bf*)(rowK0 + off + 16);
  v16bf r;
#pragma unroll
  for (int i = 0; i < 8; ++i) { r[i] = lo[i]; r[i + 8] = hi[i]; }
  return r;
}

// B-operand (32x16 bf16, KxN): lane holds one column N;
// lanes 0-15 hold K{0..7,16..23}, lanes 16-31 hold K{8..15,24..31} (matches A).
static __device__ __forceinline__ v16bf load_b(const bf16_t* colK0, int lane) {
  const int off = (lane & 16) ? 8 : 0;
  v8bf lo = *(const v8bf*)(colK0 + off);
  v8bf hi = *(const v8bf*)(colK0 + off + 16);
  v16bf r;
#pragma unroll
  for (int i = 0; i < 8; ++i) { r[i] = lo[i]; r[i + 8] = hi[i]; }
  return r;
}

static __device__ __forceinline__ v8f wmma_bf16(v16bf a, v16bf b, v8f c) {
  return __builtin_amdgcn_wmma_f32_16x16x32_bf16(false, a, false, b, (short)0, c,
                                                 false, false);
}

// ---------------------------------------------------------------- fp32 -> bf16
__global__ void __launch_bounds__(256)
cvt_f32_bf16(const float* __restrict__ src, bf16_t* __restrict__ dst, int n) {
  int i = blockIdx.x * blockDim.x + threadIdx.x;
  const int stride = gridDim.x * blockDim.x;
  for (; i < n; i += stride) dst[i] = (bf16_t)src[i];
}

// --------------------------------------------- fused QKV GEMM + bias + RoPE
// grid: (ceil(MROWS/64), 36)  y: matID = y/12 (0=q,1=k,2=v), nTile = y%12
// block: 128 threads = 4 waves, each wave: 16 rows x 64 cols.
__global__ void __launch_bounds__(128)
qkv_rope_kernel(const bf16_t* __restrict__ xbf, const bf16_t* __restrict__ wqkv,
                const float* __restrict__ qbias, const float* __restrict__ vbias,
                bf16_t* __restrict__ qout, bf16_t* __restrict__ kout,
                bf16_t* __restrict__ vtout,
                const int* __restrict__ wGrid, const int* __restrict__ pPre) {
  const int lane  = threadIdx.x & 31;
  const int wave  = threadIdx.x >> 5;
  const int matID = blockIdx.y / 12;
  const int nt    = blockIdx.y % 12;
  const int mBase = blockIdx.x * 64 + wave * 16;
  int aRow = mBase + (lane & 15);
  if (aRow > MROWS - 1) aRow = MROWS - 1;
  const bf16_t* wmat = wqkv + (size_t)matID * D_MODEL * D_MODEL;
  const bf16_t* arow = xbf + (size_t)aRow * D_MODEL;

  v8f acc[4];
#pragma unroll
  for (int s = 0; s < 4; ++s) acc[s] = vzero();

  for (int k = 0; k < D_MODEL; k += 32) {
    const v16bf a = load_a(arow + k, lane);
#pragma unroll
    for (int s = 0; s < 4; ++s) {
      const int n = nt * 64 + s * 16 + (lane & 15);
      const v16bf b = load_b(wmat + (size_t)n * D_MODEL + k, lane);
      acc[s] = wmma_bf16(a, b, acc[s]);
    }
  }

  const int gw = wGrid[0];     // w = 32
  const int p  = pPre[0];      // num_prefix_tokens = 1
  const int hb = (lane & 16) ? 8 : 0;

#pragma unroll
  for (int s = 0; s < 4; ++s) {
    const int e    = nt * 64 + s * 16 + (lane & 15);
    const int hIdx = e >> 6;
    const int d    = e & 63;
    const float bias = (matID == 0) ? qbias[e] : ((matID == 2) ? vbias[e] : 0.0f);
#pragma unroll
    for (int i = 0; i < 8; ++i) {
      const int m = mBase + hb + i;
      float val = acc[s][i] + bias;
      const float other = __shfl_xor(val, 1, 32);  // pair partner (d^1), uniform
      if (matID != 2 && m < MROWS) {
        const int sTok = m % SEQ;
        if (sTok >= p) {
          const int idx = sTok - p;
          const int yy = idx / gw;
          const int xx = idx % gw;
          const int pk = d >> 1;
          const float fr  = __powf(ROPE_THETA,
                                   -(float)(pk & (QUARTER - 1)) * (1.0f / (float)QUARTER));
          const float ang = ((pk < QUARTER) ? (float)yy : (float)xx) * fr;
          const float c  = __cosf(ang);
          const float sn = __sinf(ang);
          val = (d & 1) ? (val * c + other * sn) : (val * c - other * sn);
        }
      }
      if (matID == 0) val *= 0.125f;  // hd^-0.5 folded into q
      if (m < MROWS) {
        const int bIdx = m / SEQ;
        const int sTok = m % SEQ;
        const size_t bh = (size_t)bIdx * NHEADS + hIdx;
        const bf16_t bv = (bf16_t)val;
        if (matID == 0)      qout [(bh * SPAD + sTok) * HDIM + d] = bv;
        else if (matID == 1) kout [(bh * SPAD + sTok) * HDIM + d] = bv;
        else                 vtout[(bh * HDIM + d) * SPAD + sTok] = bv;  // transposed
      }
    }
  }
}

// ---------------------------------------- flash attention, 1 wave / 16 queries
// grid: (ceil(SEQ/16), NHEADS, BATCH), block: 32 threads.
__global__ void __launch_bounds__(32)
attn_kernel(const bf16_t* __restrict__ qbf, const bf16_t* __restrict__ kbf,
            const bf16_t* __restrict__ vtbf, bf16_t* __restrict__ ctx) {
  const int lane = threadIdx.x & 31;
  const int qt = blockIdx.x;
  const int h  = blockIdx.y;
  const int b  = blockIdx.z;
  const size_t bh = (size_t)b * NHEADS + h;
  const bf16_t* qp = qbf + bh * SPAD * HDIM;
  const bf16_t* kp = kbf + bh * SPAD * HDIM;
  const bf16_t* vp = vtbf + bh * HDIM * SPAD;

  const int mrow = qt * 16 + (lane & 15);
  const v16bf qA0 = load_a(qp + (size_t)mrow * HDIM + 0, lane);
  const v16bf qA1 = load_a(qp + (size_t)mrow * HDIM + 32, lane);

  v8f acc[4];
#pragma unroll
  for (int g = 0; g < 4; ++g) acc[g] = vzero();
  float mr[8], lr[8];
#pragma unroll
  for (int i = 0; i < 8; ++i) { mr[i] = -3.0e38f; lr[i] = 0.0f; }

  __shared__ __align__(16) bf16_t pbuf[16 * 32];
  const int hb = (lane & 16) ? 8 : 0;

  for (int kc = 0; kc < SPAD; kc += 32) {
    v8f sc[2];
#pragma unroll
    for (int t = 0; t < 2; ++t) {
      const int krow = kc + t * 16 + (lane & 15);
      const v16bf kB0 = load_b(kp + (size_t)krow * HDIM + 0, lane);
      const v16bf kB1 = load_b(kp + (size_t)krow * HDIM + 32, lane);
      v8f c = vzero();
      c = wmma_bf16(qA0, kB0, c);
      c = wmma_bf16(qA1, kB1, c);
      const int keyIdx = kc + t * 16 + (lane & 15);
      if (keyIdx >= SEQ) {
#pragma unroll
        for (int i = 0; i < 8; ++i) c[i] = -3.0e38f;   // mask padded keys
      }
      sc[t] = c;
    }
    // online softmax (rows live across 16 lanes of each half-wave)
    float al[8];
#pragma unroll
    for (int i = 0; i < 8; ++i) {
      float v = fmaxf(sc[0][i], sc[1][i]);
#pragma unroll
      for (int o = 8; o >= 1; o >>= 1) v = fmaxf(v, __shfl_xor(v, o, 16));
      const float mn = fmaxf(mr[i], v);
      al[i] = __expf(mr[i] - mn);
      mr[i] = mn;
    }
#pragma unroll
    for (int i = 0; i < 8; ++i) {
      const float p0 = __expf(sc[0][i] - mr[i]);
      const float p1 = __expf(sc[1][i] - mr[i]);
      pbuf[(hb + i) * 32 + (lane & 15)]      = (bf16_t)p0;
      pbuf[(hb + i) * 32 + 16 + (lane & 15)] = (bf16_t)p1;
      float ps = p0 + p1;
#pragma unroll
      for (int o = 8; o >= 1; o >>= 1) ps += __shfl_xor(ps, o, 16);
      lr[i] = lr[i] * al[i] + ps;
    }
    __syncthreads();
    const v16bf pA = load_a(pbuf + (lane & 15) * 32, lane);  // C-layout -> A-layout
#pragma unroll
    for (int g = 0; g < 4; ++g) {
#pragma unroll
      for (int i = 0; i < 8; ++i) acc[g][i] *= al[i];
      const int d = g * 16 + (lane & 15);
      const v16bf vB = load_b(vp + (size_t)d * SPAD + kc, lane);
      acc[g] = wmma_bf16(pA, vB, acc[g]);
    }
    __syncthreads();
  }

#pragma unroll
  for (int i = 0; i < 8; ++i) lr[i] = 1.0f / lr[i];
#pragma unroll
  for (int g = 0; g < 4; ++g) {
#pragma unroll
    for (int i = 0; i < 8; ++i) {
      const int sTok = qt * 16 + hb + i;
      if (sTok < SEQ) {
        const float o = acc[g][i] * lr[i];
        ctx[((size_t)b * SEQ + sTok) * D_MODEL + h * HDIM + g * 16 + (lane & 15)]
            = (bf16_t)o;
      }
    }
  }
}

// ----------------------------------------------- output projection, fp32 out
__global__ void __launch_bounds__(128)
oproj_kernel(const bf16_t* __restrict__ ctx, const bf16_t* __restrict__ wo,
             const float* __restrict__ ob, float* __restrict__ out) {
  const int lane = threadIdx.x & 31;
  const int wave = threadIdx.x >> 5;
  const int nt   = blockIdx.y;
  const int mBase = blockIdx.x * 64 + wave * 16;
  int aRow = mBase + (lane & 15);
  if (aRow > MROWS - 1) aRow = MROWS - 1;
  const bf16_t* arow = ctx + (size_t)aRow * D_MODEL;

  v8f acc[4];
#pragma unroll
  for (int s = 0; s < 4; ++s) acc[s] = vzero();

  for (int k = 0; k < D_MODEL; k += 32) {
    const v16bf a = load_a(arow + k, lane);
#pragma unroll
    for (int s = 0; s < 4; ++s) {
      const int n = nt * 64 + s * 16 + (lane & 15);
      const v16bf b = load_b(wo + (size_t)n * D_MODEL + k, lane);
      acc[s] = wmma_bf16(a, b, acc[s]);
    }
  }
  const int hb = (lane & 16) ? 8 : 0;
#pragma unroll
  for (int s = 0; s < 4; ++s) {
    const int n = nt * 64 + s * 16 + (lane & 15);
    const float bias = ob[n];
#pragma unroll
    for (int i = 0; i < 8; ++i) {
      const int m = mBase + hb + i;
      if (m < MROWS) out[(size_t)m * D_MODEL + n] = acc[s][i] + bias;
    }
  }
}

extern "C" void kernel_launch(void* const* d_in, const int* in_sizes, int n_in,
                              void* d_out, int out_size, void* d_ws, size_t ws_size,
                              hipStream_t stream) {
  const float* x  = (const float*)d_in[0];
  const float* qw = (const float*)d_in[1];
  const float* qb = (const float*)d_in[2];
  const float* kw = (const float*)d_in[3];
  const float* vw = (const float*)d_in[4];
  const float* vb = (const float*)d_in[5];
  const float* ow = (const float*)d_in[6];
  const float* ob = (const float*)d_in[7];
  const int* wPtr = (const int*)d_in[9];
  const int* pPtr = (const int*)d_in[10];

  char* ws = (char*)d_ws;
  size_t off = 0;
  auto take = [&](size_t elems) -> void* {
    void* p = ws + off;
    off += elems * sizeof(bf16_t);
    return p;
  };
  const size_t W_ELEMS = (size_t)D_MODEL * D_MODEL;
  const size_t QKV_ELEMS = (size_t)BATCH * NHEADS * SPAD * HDIM;
  bf16_t* xbf  = (bf16_t*)take((size_t)MROWS * D_MODEL);
  bf16_t* wqkv = (bf16_t*)take(3 * W_ELEMS);
  bf16_t* wo   = (bf16_t*)take(W_ELEMS);
  bf16_t* qbf  = (bf16_t*)take(QKV_ELEMS);
  bf16_t* kbf  = (bf16_t*)take(QKV_ELEMS);
  bf16_t* vtbf = (bf16_t*)take(QKV_ELEMS);
  bf16_t* ctx  = (bf16_t*)take((size_t)MROWS * D_MODEL);
  (void)kbf; (void)vtbf;

  // zero Q/K/Vt (contiguous) so padded rows are defined (no NaN into WMMA)
  hipMemsetAsync(qbf, 0, 3 * QKV_ELEMS * sizeof(bf16_t), stream);

  auto cvt = [&](const float* src, bf16_t* dst, int n) {
    int blocks = (n + 255) / 256;
    if (blocks > 2048) blocks = 2048;
    cvt_f32_bf16<<<blocks, 256, 0, stream>>>(src, dst, n);
  };
  cvt(x,  xbf,  MROWS * D_MODEL);
  cvt(qw, wqkv,                 (int)W_ELEMS);
  cvt(kw, wqkv + W_ELEMS,       (int)W_ELEMS);
  cvt(vw, wqkv + 2 * W_ELEMS,   (int)W_ELEMS);
  cvt(ow, wo,                   (int)W_ELEMS);

  dim3 gQ((MROWS + 63) / 64, 36);
  qkv_rope_kernel<<<gQ, 128, 0, stream>>>(xbf, wqkv, qb, vb, qbf, kbf, vtbf,
                                          wPtr, pPtr);

  dim3 gA((SEQ + 15) / 16, NHEADS, BATCH);
  attn_kernel<<<gA, 32, 0, stream>>>(qbf, kbf, vtbf, ctx);

  dim3 gO((MROWS + 63) / 64, 12);
  oproj_kernel<<<gO, 128, 0, stream>>>(ctx, wo, ob, (float*)d_out);
}
